// VGAE_13108240187475
// MI455X (gfx1250) — matrix operand order repaired and checked by reading, not compiled, not run on hardware
//
#include <hip/hip_runtime.h>
#include <hip/hip_bf16.h>

// ---------------------------------------------------------------------------
// VGAE forward on MI455X (gfx1250, wave32, bf16 WMMA, f32 accumulate)
//   hidden = relu(adj @ (X @ W0))
//   M      = adj @ (hidden @ [W_mu | W_logstd])     (mean | logstd fused)
//   Z      = noise * exp(logstd) + mean
//   A_pred = sigmoid(Z @ Z^T)
//
// B operands pre-transposed bf16 [Nc][K], staged with
// GLOBAL_LOAD_ASYNC_TO_LDS_B128 (ASYNCcnt). A (adj) converted f32->bf16 on
// the fly. Double-buffered LDS with the K-loop unrolled by 2 so each buffer
// is a compile-time constant (static ds addressing, no flat ops).
// One barrier per K-step; staging overlapped with the 16-wmma burst.
// ---------------------------------------------------------------------------

#define N_NODES 12288
#define IN_DIM  512
#define H1      256
#define H2      128

typedef unsigned short u16;
typedef __attribute__((ext_vector_type(16))) __bf16 v16bf;
typedef __attribute__((ext_vector_type(8)))  float  v8f;

struct Frag32B { uint4 x, y; };   // 32 bytes == v16bf

static __device__ inline v16bf frag_cast(uint4 x, uint4 y) {
    Frag32B f{x, y};
    return __builtin_bit_cast(v16bf, f);
}
static __device__ inline unsigned bf2pack(float lo, float hi) {
    unsigned a = __builtin_bit_cast(unsigned, lo);
    unsigned b = __builtin_bit_cast(unsigned, hi);
    a = (a + 0x8000u) >> 16;
    b = (b + 0x8000u) >> 16;
    return a | (b << 16);
}
static __device__ inline u16 f2bf(float f) {
    unsigned a = __builtin_bit_cast(unsigned, f);
    return (u16)((a + 0x8000u) >> 16);
}

enum { EPI_NONE = 0, EPI_RELU = 1, EPI_SIGMOID = 2 };
enum { OUT_F32 = 0, OUT_TBF16 = 1 };   // OUT_TBF16: C^T bf16, leading dim LDC = M

template <int EPI>
static __device__ inline float epi_apply(float v) {
    if (EPI == EPI_RELU)    return fmaxf(v, 0.0f);
    if (EPI == EPI_SIGMOID) return 1.0f / (1.0f + __expf(-v));
    return v;
}

// 4x async 16B/lane copies: global row slice -> this thread's LDS B row
static __device__ inline void async_b_tile(const u16* src, unsigned sbaddr) {
    asm volatile("global_load_async_to_lds_b128 %0, %1, off"
                 :: "v"(sbaddr), "v"(src) : "memory");
    asm volatile("global_load_async_to_lds_b128 %0, %1, off offset:16"
                 :: "v"(sbaddr), "v"(src) : "memory");
    asm volatile("global_load_async_to_lds_b128 %0, %1, off offset:32"
                 :: "v"(sbaddr), "v"(src) : "memory");
    asm volatile("global_load_async_to_lds_b128 %0, %1, off offset:48"
                 :: "v"(sbaddr), "v"(src) : "memory");
}

struct AF { float4 f0, f1, f2, f3; };
static __device__ inline AF load_a(const float* src) {
    AF a;
    a.f0 = *(const float4*)(src + 0);
    a.f1 = *(const float4*)(src + 4);
    a.f2 = *(const float4*)(src + 8);
    a.f3 = *(const float4*)(src + 12);
    return a;
}
static __device__ inline void store_a(const AF& a, u16* sa) {
    uint4 p0 = { bf2pack(a.f0.x, a.f0.y), bf2pack(a.f0.z, a.f0.w),
                 bf2pack(a.f1.x, a.f1.y), bf2pack(a.f1.z, a.f1.w) };
    uint4 p1 = { bf2pack(a.f2.x, a.f2.y), bf2pack(a.f2.z, a.f2.w),
                 bf2pack(a.f3.x, a.f3.y), bf2pack(a.f3.z, a.f3.w) };
    *(uint4*)(sa + 0) = p0;
    *(uint4*)(sa + 8) = p1;
}

// ---------------------------------------------------------------------------
// C[M,Nc] = epi( A[M,K](f32) * B )  with Bt = B^T as bf16 [Nc][K] in memory.
// Block tile: 128 rows x 256 cols, K-step 32 (loop unrolled x2),
// 256 threads = 8 waves (4x2); each wave: 32x128 = 2x8 WMMA accumulators.
// Requires K % 64 == 0.
// ---------------------------------------------------------------------------
template <int EPI, int OUT, int LDC>
__global__ __launch_bounds__(256)
void gemm_wmma(const float* __restrict__ A, int lda,
               const u16* __restrict__ Bt, int K,
               void* __restrict__ Cv)
{
    constexpr int LDSS  = 40;                   // LDS row stride in halves
    constexpr int ASZ   = 128 * LDSS;           // halves per A buffer
    constexpr int BSZ   = 256 * LDSS;           // halves per B buffer
    __shared__ u16 lds_a[2 * ASZ];              // [buf][row][k] bf16
    __shared__ u16 lds_b[2 * BSZ];              // [buf][col][k] bf16

    const int tid  = threadIdx.x;
    const int lane = tid & 31;
    const int wid  = tid >> 5;
    const int wm   = wid & 3;                   // 0..3 -> 32-row slab
    const int wn   = wid >> 2;                  // 0..1 -> 128-col slab
    const int lr   = lane & 15;
    const int lh   = lane >> 4;

    const int m0 = blockIdx.y * 128;
    const int n0 = blockIdx.x * 256;

    const float* arow = A + (size_t)(m0 + (tid >> 1)) * lda + (tid & 1) * 16;
    const u16*   brow = Bt + (size_t)(n0 + tid) * K;

    const int sa_idx = (tid >> 1) * LDSS + (tid & 1) * 16;
    const unsigned sb0 = (unsigned)(uintptr_t)&lds_b[tid * LDSS];        // buf 0
    const unsigned sb1 = (unsigned)(uintptr_t)&lds_b[BSZ + tid * LDSS]; // buf 1

    v8f acc[2][8] = {};

    const int abase = wm * 32 * LDSS + lh * 8;   // + mi*16*LDSS
    const int bbase = wn * 128 * LDSS + lh * 8;  // + ni*16*LDSS
    const int lroff = lr * LDSS;

    // one pipeline step: compute from CUR buffer, stage k_next into NXT
    auto kstep = [&](int k_next, bool has_next, const u16* la, const u16* lb,
                     u16* sa_nxt, unsigned sb_nxt) {
        AF an{};
        if (has_next) {
            async_b_tile(brow + k_next, sb_nxt);
            __builtin_prefetch(arow + k_next + 32, 0, 3);  // speculative
            an = load_a(arow + k_next);
        }
        v16bf afr[2];
#pragma unroll
        for (int mi = 0; mi < 2; ++mi) {
            const int base = abase + mi * 16 * LDSS + lroff;
            afr[mi] = frag_cast(*(const uint4*)&la[base],
                                *(const uint4*)&la[base + 16]);
        }
#pragma unroll
        for (int ni = 0; ni < 8; ++ni) {
            const int base = bbase + ni * 16 * LDSS + lroff;
            v16bf bfr = frag_cast(*(const uint4*)&lb[base],
                                  *(const uint4*)&lb[base + 16]);
            acc[0][ni] = __builtin_amdgcn_wmma_f32_16x16x32_bf16(
                false, afr[0], false, bfr, (short)0, acc[0][ni], false, false);
            acc[1][ni] = __builtin_amdgcn_wmma_f32_16x16x32_bf16(
                false, afr[1], false, bfr, (short)0, acc[1][ni], false, false);
        }
        if (has_next) store_a(an, sa_nxt);
        asm volatile("s_wait_asynccnt 0x0" ::: "memory");
        __syncthreads();
    };

    // ---- prologue: stage tile 0 into buffer 0 -------------------------
    async_b_tile(brow, sb0);
    store_a(load_a(arow), &lds_a[sa_idx]);
    asm volatile("s_wait_asynccnt 0x0" ::: "memory");
    __syncthreads();

    // ---- main loop, unrolled x2 so buffers are compile-time -----------
    for (int k0 = 0; k0 < K; k0 += 64) {
        // compute buf0 (tile k0), stage k0+32 into buf1 (always exists)
        kstep(k0 + 32, true,
              &lds_a[0], &lds_b[0], &lds_a[ASZ + sa_idx], sb1);
        // compute buf1 (tile k0+32), stage k0+64 into buf0 (if any)
        kstep(k0 + 64, (k0 + 64) < K,
              &lds_a[ASZ], &lds_b[BSZ], &lds_a[sa_idx], sb0);
    }

    // ---- epilogue -----------------------------------------------------
#pragma unroll
    for (int mi = 0; mi < 2; ++mi) {
#pragma unroll
        for (int ni = 0; ni < 8; ++ni) {
            const int row0 = m0 + wm * 32 + mi * 16 + lh * 8;
            const int col  = n0 + wn * 128 + ni * 16 + lr;
            if (OUT == OUT_F32) {
                float* C = (float*)Cv + (row0 * LDC + col);   // 32-bit offset
#pragma unroll
                for (int r = 0; r < 8; ++r)                   // r*LDC -> imm offs
                    C[r * LDC] = epi_apply<EPI>(acc[mi][ni][r]);
            } else {
                // transposed bf16: 8 consecutive rows per lane -> one b128
                float v0 = epi_apply<EPI>(acc[mi][ni][0]);
                float v1 = epi_apply<EPI>(acc[mi][ni][1]);
                float v2 = epi_apply<EPI>(acc[mi][ni][2]);
                float v3 = epi_apply<EPI>(acc[mi][ni][3]);
                float v4 = epi_apply<EPI>(acc[mi][ni][4]);
                float v5 = epi_apply<EPI>(acc[mi][ni][5]);
                float v6 = epi_apply<EPI>(acc[mi][ni][6]);
                float v7 = epi_apply<EPI>(acc[mi][ni][7]);
                uint4 p = { bf2pack(v0, v1), bf2pack(v2, v3),
                            bf2pack(v4, v5), bf2pack(v6, v7) };
                u16* Ct = (u16*)Cv + (col * LDC + row0);      // 32-bit offset
                *(uint4*)Ct = p;
            }
        }
    }
}

// W0t[256][512] bf16 = W0[512][256]^T
__global__ void build_w0t(const float* __restrict__ W0, u16* __restrict__ W0t)
{
    int i = blockIdx.x * blockDim.x + threadIdx.x;   // over 512*256
    if (i < IN_DIM * H1) {
        int k = i >> 8, c = i & 255;
        W0t[c * IN_DIM + k] = f2bf(W0[i]);
    }
}

// WcatT[256][256] bf16 = [W_mu | W_logstd]^T
__global__ void build_wcat_t(const float* __restrict__ wmu,
                             const float* __restrict__ wls,
                             u16* __restrict__ wcatT)
{
    int i = blockIdx.x * blockDim.x + threadIdx.x;   // over 256*256
    if (i < H1 * 256) {
        int k = i >> 8, c = i & 255;
        float v = (c < H2) ? wmu[k * H2 + c] : wls[k * H2 + (c - H2)];
        wcatT[c * H1 + k] = f2bf(v);
    }
}

// Z = noise * exp(logstd) + mean ; writes f32 (A-side) and bf16 (B-side)
__global__ void reparam(const float* __restrict__ Mbuf,
                        const float* __restrict__ noise,
                        float* __restrict__ Zf, u16* __restrict__ Zb)
{
    int i = blockIdx.x * blockDim.x + threadIdx.x;   // over N*128
    if (i < N_NODES * H2) {
        int r = i >> 7, c = i & 127;
        float mean = Mbuf[r * 256 + c];
        float ls   = Mbuf[r * 256 + 128 + c];
        float z = noise[i] * __expf(ls) + mean;
        Zf[i] = z;
        Zb[i] = f2bf(z);
    }
}

extern "C" void kernel_launch(void* const* d_in, const int* in_sizes, int n_in,
                              void* d_out, int out_size, void* d_ws, size_t ws_size,
                              hipStream_t stream)
{
    const float* X    = (const float*)d_in[0];   // [N, 512]
    const float* adj  = (const float*)d_in[1];   // [N, N]
    const float* nois = (const float*)d_in[2];   // [N, 128]
    const float* W0   = (const float*)d_in[3];   // [512, 256]
    const float* Wmu  = (const float*)d_in[4];   // [256, 128]
    const float* Wls  = (const float*)d_in[5];   // [256, 128]
    float* out = (float*)d_out;                  // [N, N]

    // workspace layout (bytes), lifetimes aliased:
    //   bufT (bf16, [256][N]) : XW0t -> headst
    //   bufF (f32,  [N][256]) : hidden -> Mbuf
    char* w = (char*)d_ws;
    u16*   bufT  = (u16*)(w);                                  // 6 MB
    float* bufF  = (float*)(w + 6291456);                      // 12 MB
    float* Zf    = (float*)(w + 6291456 + 12582912);           // 6 MB
    u16*   Zb    = (u16*)(w + 6291456 + 12582912 + 6291456);   // 3 MB
    u16*   W0t   = (u16*)(w + 28311552);                       // 256 KB
    u16*   wcatT = (u16*)(w + 28311552 + 262144);              // 128 KB

    dim3 blk(256);

    build_w0t<<<dim3((IN_DIM * H1) / 256), blk, 0, stream>>>(W0, W0t);
    build_wcat_t<<<dim3((H1 * 256) / 256), blk, 0, stream>>>(Wmu, Wls, wcatT);

    // XW0t = (X @ W0)^T               bf16 [256][N]
    gemm_wmma<EPI_NONE, OUT_TBF16, N_NODES>
        <<<dim3(1, N_NODES / 128), blk, 0, stream>>>(X, IN_DIM, W0t, IN_DIM, bufT);

    // hidden = relu(adj @ XW0)        f32 [N][256]
    gemm_wmma<EPI_RELU, OUT_F32, 256>
        <<<dim3(1, N_NODES / 128), blk, 0, stream>>>(adj, N_NODES, bufT, N_NODES, bufF);

    // headst = (hidden @ Wcat)^T      bf16 [256][N]
    gemm_wmma<EPI_NONE, OUT_TBF16, N_NODES>
        <<<dim3(1, N_NODES / 128), blk, 0, stream>>>(bufF, 256, wcatT, H1, bufT);

    // Mbuf = adj @ heads              f32 [N][256] = [mean | logstd]
    gemm_wmma<EPI_NONE, OUT_F32, 256>
        <<<dim3(1, N_NODES / 128), blk, 0, stream>>>(adj, N_NODES, bufT, N_NODES, bufF);

    // Z = noise * exp(logstd) + mean  (f32 + bf16 copies)
    reparam<<<dim3((N_NODES * H2) / 256), blk, 0, stream>>>(bufF, nois, Zf, Zb);

    // A_pred = sigmoid(Z @ Z^T)       f32 [N][N]
    gemm_wmma<EPI_SIGMOID, OUT_F32, N_NODES>
        <<<dim3(N_NODES / 256, N_NODES / 128), blk, 0, stream>>>(Zf, H2, Zb, H2, out);
}
